// GraphConvolutionalNetwork_27702539059471
// MI455X (gfx1250) — compile-verified, matrix operand order checked by reading, not run on hardware
//
#include <hip/hip_runtime.h>
#include <hip/hip_bf16.h>

typedef __attribute__((ext_vector_type(16))) __bf16 v16bf;
typedef __attribute__((ext_vector_type(2)))  __bf16 v2bf;
typedef __attribute__((ext_vector_type(8)))  float  v8f;

#define D 128
#define NGRAPHS 512
#define NPACK (4 * 8 * 8 * 32)   // 8192 packed bf16-pairs = 32 KB

// ---------------------------------------------------------------- utility
__global__ void k_fill(float* __restrict__ p, float v, long n) {
    long i = (long)blockIdx.x * blockDim.x + threadIdx.x;
    if (i < n) p[i] = v;
}

__global__ void k_deg(const int* __restrict__ dst, float* __restrict__ deg, long nE) {
    long e = (long)blockIdx.x * blockDim.x + threadIdx.x;
    if (e < nE) atomicAdd(&deg[dst[e]], 1.0f);
}

__global__ void k_rsqrt(float* __restrict__ deg, long n) {
    long i = (long)blockIdx.x * blockDim.x + threadIdx.x;
    if (i < n) deg[i] = rsqrtf(deg[i]);
}

// ---------------------------------------------------------------- W pre-pack (once per layer):
// fp32 W[k][n] -> bf16 pairs in B-fragment lane order:
//   entry ((kt*8 + nt)*8 + j)*32 + lane  = (W[k][n], W[k+1][n])
__global__ void k_pack_w(const float* __restrict__ W, v2bf* __restrict__ wpk) {
    int i = blockIdx.x * blockDim.x + threadIdx.x;
    if (i >= NPACK) return;
    int lane_ = i & 31;
    int j     = (i >> 5) & 7;
    int nt    = (i >> 8) & 7;
    int kt    = i >> 11;
    int khalf = (lane_ >> 4) * 8;
    int n     = nt * 16 + (lane_ & 15);
    int k     = kt * 32 + ((j >> 2) << 4) + khalf + ((j & 3) << 1);
    v2bf p;
    p.x = (__bf16)W[k * D + n];
    p.y = (__bf16)W[(k + 1) * D + n];
    wpk[i] = p;
}

// ---------------------------------------------------------------- GEMM: t = (reluIn? relu(in) : in) @ W
// Persistent blocks, grid-stride over 64-row strips. 128 threads = 4 waves;
// each wave computes a 16x128 strip with 8 tiles of v_wmma_f32_16x16x32_bf16.
// Pre-packed W is copied into LDS once per block (raw uint4 copies).
__global__ void __launch_bounds__(128) k_gemm_wmma(
    const float* __restrict__ in, const v2bf* __restrict__ wpk,
    float* __restrict__ t, int nNodes, int nStrips, int reluIn)
{
    __shared__ v2bf wp[NPACK];     // 32 KB

    const int tid = threadIdx.x;
    {
        const uint4* __restrict__ s4 = (const uint4*)wpk;
        uint4* d4 = (uint4*)wp;
        for (int i = tid; i < NPACK / 4; i += 128) d4[i] = s4[i];
    }
    __syncthreads();

    const int lane  = tid & 31;
    const int wave  = tid >> 5;
    const int m     = lane & 15;
    const int khalf = (lane >> 4) * 8;
    const int nlo   = lane & 15;
    const int mAdd  = (lane >> 4) * 8;

    for (int strip = blockIdx.x; strip < nStrips; strip += gridDim.x) {
        const int rowBase = strip * 64 + wave * 16;
        int row = rowBase + m;
        if (row >= nNodes) row = nNodes - 1;       // keep EXEC all-ones; store is guarded
        const float* __restrict__ arow = in + (long)row * D;

        v8f c[8];
#pragma unroll
        for (int nt = 0; nt < 8; ++nt) {
            v8f z = {0.f, 0.f, 0.f, 0.f, 0.f, 0.f, 0.f, 0.f};
            c[nt] = z;
        }

#pragma unroll
        for (int kt = 0; kt < 4; ++kt) {
            // A fragment: 16x32 bf16, row m, K pairs per VGPR
            v16bf a;
#pragma unroll
            for (int j = 0; j < 8; ++j) {
                int k = kt * 32 + ((j >> 2) << 4) + khalf + ((j & 3) << 1);
                float f0 = arow[k];
                float f1 = arow[k + 1];
                if (reluIn) { f0 = fmaxf(f0, 0.f); f1 = fmaxf(f1, 0.f); }
                a[2 * j]     = (__bf16)f0;
                a[2 * j + 1] = (__bf16)f1;
            }
#pragma unroll
            for (int nt = 0; nt < 8; ++nt) {
                v16bf b;
                const int base = ((kt * 8 + nt) * 8) * 32 + lane;
#pragma unroll
                for (int j = 0; j < 8; ++j) {
                    v2bf p = wp[base + j * 32];
                    b[2 * j]     = p.x;
                    b[2 * j + 1] = p.y;
                }
                c[nt] = __builtin_amdgcn_wmma_f32_16x16x32_bf16(
                    false, a, false, b, (short)0, c[nt], false, false);
            }
        }

        // store: VGPR v -> M = v + (lane>=16 ? 8 : 0), N = nt*16 + (lane&15)
#pragma unroll
        for (int nt = 0; nt < 8; ++nt) {
#pragma unroll
            for (int v = 0; v < 8; ++v) {
                int mm = rowBase + v + mAdd;
                if (mm < nNodes) t[(long)mm * D + nt * 16 + nlo] = c[nt][v];
            }
        }
    }
}

// ---------------------------------------------------------------- acc = t * snorm + bias (initializes accumulator)
__global__ void k_self_bias(const float* __restrict__ t, const float* __restrict__ dis,
                            const float* __restrict__ bias, float* __restrict__ acc, int nNodes)
{
    long i = (long)blockIdx.x * blockDim.x + threadIdx.x;
    if (i >= (long)nNodes * D) return;
    int node = (int)(i >> 7);
    int col  = (int)(i & (D - 1));
    float dn = dis[node];
    acc[i] = t[i] * (dn * dn) + bias[col];
}

// ---------------------------------------------------------------- acc[dst] += t[src] * (dis[src]*dis[dst]); 32 threads/edge, float4 each
__global__ void k_scatter(const float* __restrict__ t, const int* __restrict__ src,
                          const int* __restrict__ dst, const float* __restrict__ dis,
                          float* __restrict__ acc, long nE)
{
    long tix = (long)blockIdx.x * blockDim.x + threadIdx.x;
    long e = tix >> 5;
    int  chunk = (int)(tix & 31);
    if (e >= nE) return;
    int s = src[e], d = dst[e];
    float w = dis[s] * dis[d];
    const float4 v = *(const float4*)(t + (long)s * D + chunk * 4);
    float* p = acc + (long)d * D + chunk * 4;
    atomicAdd(p + 0, v.x * w);
    atomicAdd(p + 1, v.y * w);
    atomicAdd(p + 2, v.z * w);
    atomicAdd(p + 3, v.w * w);
}

// ---------------------------------------------------------------- pooling: segment mean over batch (relu applied on read)
__global__ void k_pool_accum(const float* __restrict__ h, const int* __restrict__ batch,
                             float* __restrict__ pooled, float* __restrict__ counts, int nNodes)
{
    long tix = (long)blockIdx.x * blockDim.x + threadIdx.x;
    long node = tix >> 5;
    int  chunk = (int)(tix & 31);
    if (node >= nNodes) return;
    int g = batch[node];
    const float4 v = *(const float4*)(h + node * D + chunk * 4);
    float* p = pooled + (long)g * D + chunk * 4;
    atomicAdd(p + 0, fmaxf(v.x, 0.f));
    atomicAdd(p + 1, fmaxf(v.y, 0.f));
    atomicAdd(p + 2, fmaxf(v.z, 0.f));
    atomicAdd(p + 3, fmaxf(v.w, 0.f));
    if (chunk == 0) atomicAdd(&counts[g], 1.0f);
}

__global__ void __launch_bounds__(128) k_pool_final(
    const float* __restrict__ pooled, const float* __restrict__ counts,
    const float* __restrict__ Wfc, const float* __restrict__ bfc, float* __restrict__ out)
{
    __shared__ float red[D];
    int g = blockIdx.x, tid = threadIdx.x;
    float cnt = fmaxf(counts[g], 1.0f);
    red[tid] = (pooled[(long)g * D + tid] / cnt) * Wfc[tid];
    __syncthreads();
    for (int st = 64; st > 0; st >>= 1) {
        if (tid < st) red[tid] += red[tid + st];
        __syncthreads();
    }
    if (tid == 0) out[g] = red[0] + bfc[0];
}

// ---------------------------------------------------------------- launch
extern "C" void kernel_launch(void* const* d_in, const int* in_sizes, int n_in,
                              void* d_out, int out_size, void* d_ws, size_t ws_size,
                              hipStream_t stream)
{
    const float* x    = (const float*)d_in[0];
    const int*   ei   = (const int*)d_in[1];
    const int*   batch= (const int*)d_in[2];
    const float* W1   = (const float*)d_in[3];
    const float* b1   = (const float*)d_in[4];
    const float* W2   = (const float*)d_in[5];
    const float* b2   = (const float*)d_in[6];
    const float* W3   = (const float*)d_in[7];
    const float* b3   = (const float*)d_in[8];
    const float* Wfc  = (const float*)d_in[9];
    const float* bfc  = (const float*)d_in[10];
    float* out = (float*)d_out;

    const int  N = in_sizes[0] / D;
    const long E = (long)in_sizes[1] / 2;
    const int* src = ei;
    const int* dst = ei + E;

    // workspace carve (floats): dis[N] | t[N*D] | h[N*D] | pooled[G*D] | counts[G] | wpk[NPACK pairs]
    float* ws     = (float*)d_ws;
    float* dis    = ws;
    float* t      = dis + N;
    float* h      = t + (size_t)N * D;
    float* pooled = h + (size_t)N * D;
    float* counts = pooled + (size_t)NGRAPHS * D;
    v2bf*  wpk    = (v2bf*)(counts + NGRAPHS);

    const int  TB = 256;
    const long ND = (long)N * D;
    const unsigned gN   = (unsigned)((N + TB - 1) / TB);
    const unsigned gE   = (unsigned)((E + TB - 1) / TB);
    const unsigned gND  = (unsigned)((ND + TB - 1) / TB);
    const unsigned gE32 = (unsigned)((E * 32 + TB - 1) / TB);
    const unsigned gN32 = (unsigned)(((long)N * 32 + TB - 1) / TB);

    const int nStrips = (N + 63) / 64;
    unsigned gGem = (unsigned)nStrips;
    if (gGem > 1024u) gGem = 1024u;            // persistent blocks, grid-stride
    const unsigned gPack = (NPACK + TB - 1) / TB;

    // degree -> dis = rsqrt(indeg + 1)
    k_fill  <<<gN, TB, 0, stream>>>(dis, 1.0f, (long)N);
    k_deg   <<<gE, TB, 0, stream>>>(dst, dis, E);
    k_rsqrt <<<gN, TB, 0, stream>>>(dis, (long)N);

    // layer 1: t = x @ W1 ; h = t*snorm + b1 ; h += scatter(t*enorm)
    k_pack_w    <<<gPack, TB, 0, stream>>>(W1, wpk);
    k_gemm_wmma <<<gGem, 128, 0, stream>>>(x, wpk, t, N, nStrips, 0);
    k_self_bias <<<gND,  TB,  0, stream>>>(t, dis, b1, h, N);
    k_scatter   <<<gE32, TB,  0, stream>>>(t, src, dst, dis, h, E);

    // layer 2 (relu of previous output folded into A-fragment load)
    k_pack_w    <<<gPack, TB, 0, stream>>>(W2, wpk);
    k_gemm_wmma <<<gGem, 128, 0, stream>>>(h, wpk, t, N, nStrips, 1);
    k_self_bias <<<gND,  TB,  0, stream>>>(t, dis, b2, h, N);
    k_scatter   <<<gE32, TB,  0, stream>>>(t, src, dst, dis, h, E);

    // layer 3
    k_pack_w    <<<gPack, TB, 0, stream>>>(W3, wpk);
    k_gemm_wmma <<<gGem, 128, 0, stream>>>(h, wpk, t, N, nStrips, 1);
    k_self_bias <<<gND,  TB,  0, stream>>>(t, dis, b3, h, N);
    k_scatter   <<<gE32, TB,  0, stream>>>(t, src, dst, dis, h, E);

    // global mean pool (relu folded into read) + FC
    k_fill <<<(NGRAPHS * D + TB - 1) / TB, TB, 0, stream>>>(pooled, 0.0f, (long)NGRAPHS * D);
    k_fill <<<(NGRAPHS + TB - 1) / TB,     TB, 0, stream>>>(counts, 0.0f, (long)NGRAPHS);
    k_pool_accum <<<gN32, TB, 0, stream>>>(h, batch, pooled, counts, N);
    k_pool_final <<<NGRAPHS, 128, 0, stream>>>(pooled, counts, Wfc, bfc, out);
}
